// Model_32573031973602
// MI455X (gfx1250) — matrix operand order repaired and checked by reading, not compiled, not run on hardware
//
#include <hip/hip_runtime.h>
#include <math.h>

typedef _Float16 f16;
typedef __attribute__((ext_vector_type(16))) _Float16 v16h;
typedef __attribute__((ext_vector_type(8)))  _Float16 v8h;
typedef __attribute__((ext_vector_type(8)))  float    v8f;

#define BATCH   64
#define SEQLEN  2048
#define CIN     8
#define PRED    96
#define PLEN    16
#define PNUM    128
#define DMODEL  768
#define NHEAD   12
#define DHEAD   64
#define DFFN    3072
#define NLAYER  4
#define NEXO    7
#define EPSV    1e-5f

#define GF_BIAS 1
#define GF_GELU 2
#define GF_RELU 4
#define GF_RES  8
#define GF_O32  16
#define GF_O16  32

// ---------------------------------------------------------------- CDNA5 helpers
__device__ __forceinline__ v8f wmma_f16(v16h a, v16h b, v8f c) {
  return __builtin_amdgcn_wmma_f32_16x16x32_f16(false, a, false, b, (short)0, c,
                                                false, false);
}

// Async DMA: global memory -> LDS, 16 bytes per lane, tracked by ASYNCcnt.
// LDS byte offset = low 32 bits of the flat pointer (aperture tag is [63:32]).
__device__ __forceinline__ void async_ld_b128(void* lds, const void* g) {
  unsigned l = (unsigned)(unsigned long long)lds;
  unsigned long long ga = (unsigned long long)g;
  asm volatile("global_load_async_to_lds_b128 %0, %1, off"
               :: "v"(l), "v"(ga) : "memory");
}
__device__ __forceinline__ void wait_async_0() {
  asm volatile("s_wait_asynccnt 0x0" ::: "memory");
}
__device__ __forceinline__ void wait_async_3() {
  asm volatile("s_wait_asynccnt 0x3" ::: "memory");
}

// A-fragment (16x32 f16, MxK). Lane L (hi=L>>4, m=L&15): halves 0..7 are
// K = kb+hi*8 .. +7, halves 8..15 are K = kb+16+hi*8 .. +7 (ISA 7.12.2).
__device__ __forceinline__ v16h ld_afrag(const f16* base, int ld, int m, int hi, int kb) {
  const f16* p = base + m * ld + kb + hi * 8;
  v8h lo = *(const v8h*)p;
  v8h up = *(const v8h*)(p + 16);
  v16h r;
#pragma unroll
  for (int i = 0; i < 8; i++) { r[i] = lo[i]; r[i + 8] = up[i]; }
  return r;
}
// B-fragment (32x16 f16, KxN) from LDS stored [n][k]: lane L (n=L&15, hi=L>>4)
// halves 0..7 are K = kb+hi*8 .. +7, halves 8..15 are K = kb+16+hi*8 .. +7.
__device__ __forceinline__ v16h ld_bfrag(const f16* base, int ld, int n, int hi, int kb) {
  const f16* p = base + n * ld + kb + hi * 8;
  v8h lo = *(const v8h*)p;
  v8h up = *(const v8h*)(p + 16);
  v16h r;
#pragma unroll
  for (int i = 0; i < 8; i++) { r[i] = lo[i]; r[i + 8] = up[i]; }
  return r;
}

// ---------------------------------------------------------------- GEMM
// out[M,N] = epilogue(A[M,K](f16,rowmajor) * W[N,K]^T + bias) (+resid)
// GUARD=false: M%128==0, N%64==0 -> branch-free K loop, async double-buffered LDS.
#define ATILE (128 * 40)
#define WTILE (64 * 40)
template <bool GUARD>
__global__ __launch_bounds__(256) void gemm_wmma_kernel(
    const f16* __restrict__ A, const f16* __restrict__ W,
    const float* __restrict__ bias, const float* __restrict__ resid,
    float* __restrict__ out32, f16* __restrict__ out16,
    int M, int N, int K, int flags) {
  __shared__ __align__(16) f16 As[2 * ATILE];
  __shared__ __align__(16) f16 Ws[2 * WTILE];
  const int t = threadIdx.x;
  const int tileM = blockIdx.y * 128, tileN = blockIdx.x * 64;
  const int w = t >> 5, lane = t & 31;
  const int wm = w >> 1, wn = w & 1;
  const int hi = lane >> 4, ln = lane & 15;

  v8f acc[2][2] = {};

  if (!GUARD) {
    // per-thread staging roles (3 async b128 per thread per K-slab)
    const int ra0 = t >> 2, ka = (t & 3) * 8;   // A rows 0..63
    const int ra1 = 64 + ra0;                   // A rows 64..127
    const f16* Ag0 = A + (size_t)(tileM + ra0) * K + ka;
    const f16* Ag1 = A + (size_t)(tileM + ra1) * K + ka;
    const f16* Wg  = W + (size_t)(tileN + ra0) * K + ka;
    f16* Al0[2] = {As + ra0 * 40 + ka, As + ATILE + ra0 * 40 + ka};
    f16* Al1[2] = {As + ra1 * 40 + ka, As + ATILE + ra1 * 40 + ka};
    f16* Wl[2]  = {Ws + ra0 * 40 + ka, Ws + WTILE + ra0 * 40 + ka};

    // prologue: issue slab 0 into buffer 0
    async_ld_b128(Al0[0], Ag0);
    async_ld_b128(Al1[0], Ag1);
    async_ld_b128(Wl[0], Wg);

    const int nk = K >> 5;
    for (int ik = 0; ik < nk; ik++) {
      const int buf = ik & 1;
      if (ik + 1 < nk) {
        const int kb = (ik + 1) * 32;
        async_ld_b128(Al0[buf ^ 1], Ag0 + kb);
        async_ld_b128(Al1[buf ^ 1], Ag1 + kb);
        async_ld_b128(Wl[buf ^ 1], Wg + kb);
        wait_async_3();  // slab ik landed; slab ik+1 in flight
      } else {
        wait_async_0();
      }
      __syncthreads();
      const f16* Ab = As + buf * ATILE;
      const f16* Wb = Ws + buf * WTILE;
      v16h af[2], bfv[2];
#pragma unroll
      for (int tm = 0; tm < 2; tm++) af[tm] = ld_afrag(Ab, 40, wm * 32 + tm * 16 + ln, hi, 0);
#pragma unroll
      for (int tn = 0; tn < 2; tn++) bfv[tn] = ld_bfrag(Wb, 40, wn * 32 + tn * 16 + ln, hi, 0);
#pragma unroll
      for (int tm = 0; tm < 2; tm++)
#pragma unroll
        for (int tn = 0; tn < 2; tn++)
          acc[tm][tn] = wmma_f16(af[tm], bfv[tn], acc[tm][tn]);
      __syncthreads();  // compute done before this buffer is refilled
    }
  } else {
    // bounds-checked synchronous path (only used for tiny M=64 GEMMs)
    for (int kb = 0; kb < K; kb += 32) {
      __syncthreads();
#pragma unroll
      for (int c = t; c < 512; c += 256) {
        int r = c >> 2, ko = (c & 3) * 8;
        int gm = tileM + r;
        v8h val;
#pragma unroll
        for (int i = 0; i < 8; i++) val[i] = (f16)0.0f;
        if (gm < M) val = *(const v8h*)(A + (size_t)gm * K + kb + ko);
        *(v8h*)(As + r * 40 + ko) = val;
      }
      {
        int r = t >> 2, ko = (t & 3) * 8;
        int gn = tileN + r;
        v8h val;
#pragma unroll
        for (int i = 0; i < 8; i++) val[i] = (f16)0.0f;
        if (gn < N) val = *(const v8h*)(W + (size_t)gn * K + kb + ko);
        *(v8h*)(Ws + r * 40 + ko) = val;
      }
      __syncthreads();
      v16h af[2], bfv[2];
#pragma unroll
      for (int tm = 0; tm < 2; tm++) af[tm] = ld_afrag(As, 40, wm * 32 + tm * 16 + ln, hi, 0);
#pragma unroll
      for (int tn = 0; tn < 2; tn++) bfv[tn] = ld_bfrag(Ws, 40, wn * 32 + tn * 16 + ln, hi, 0);
#pragma unroll
      for (int tm = 0; tm < 2; tm++)
#pragma unroll
        for (int tn = 0; tn < 2; tn++)
          acc[tm][tn] = wmma_f16(af[tm], bfv[tn], acc[tm][tn]);
    }
  }

  // epilogue
#pragma unroll
  for (int tm = 0; tm < 2; tm++)
#pragma unroll
    for (int tn = 0; tn < 2; tn++) {
      int n = tileN + wn * 32 + tn * 16 + ln;
      float bv = ((flags & GF_BIAS) && (!GUARD || n < N)) ? bias[n] : 0.0f;
#pragma unroll
      for (int r = 0; r < 8; r++) {
        int m = tileM + wm * 32 + tm * 16 + hi * 8 + r;
        if (!GUARD || (m < M && n < N)) {
          float v = acc[tm][tn][r] + bv;
          if (flags & GF_GELU) v = 0.5f * v * (1.0f + erff(v * 0.70710678118f));
          if (flags & GF_RELU) v = v > 0.0f ? v : 0.0f;
          if (flags & GF_RES) v += resid[(size_t)m * N + n];
          if (flags & GF_O32) out32[(size_t)m * N + n] = v;
          if (flags & GF_O16) out16[(size_t)m * N + n] = (f16)v;
        }
      }
    }
}

// ---------------------------------------------------------------- attention
// one workgroup per (h=blockIdx.x, b=blockIdx.y); 128 queries resident,
// flash-style over S in chunks of 64. scores -> (s*tau+delta)/8 -> softmax.
#define ATTN_SMEM 90624
__global__ __launch_bounds__(256) void attn_wmma_kernel(
    const f16* __restrict__ Q, const f16* __restrict__ Kv, const f16* __restrict__ Vv,
    f16* __restrict__ O, const float* __restrict__ tau, const float* __restrict__ delta,
    int S) {
  extern __shared__ __align__(16) char smem[];
  f16* Qs = (f16*)smem;            // 128 x 72
  f16* Ks = Qs + 128 * 72;         // 64 x 72  ([s][d])
  f16* Vt = Ks + 64 * 72;          // 64 x 72  ([d][s])
  f16* Ps = Vt + 64 * 72;          // 128 x 72 probs f16
  float* Sc = (float*)(Ps + 128 * 72);  // 128 x 66 scores f32
  float* mi = Sc + 128 * 66;
  float* li = mi + 128;
  float* cr = li + 128;

  const int b = blockIdx.y, h = blockIdx.x;
  const int t = threadIdx.x, w = t >> 5, lane = t & 31;
  const int wm = w >> 1, wn = w & 1;
  const int hi = lane >> 4, ln = lane & 15;
  const float taub = tau ? tau[b] : 1.0f;
  const float delb = delta ? delta[b] : 0.0f;
  const float scale = 0.125f;  // 1/sqrt(64)

  // async preload of the whole Q block into LDS
  const f16* Qb = Q + ((size_t)b * PNUM) * DMODEL + h * DHEAD;
  for (int c = t; c < 128 * 8; c += 256) {
    int r = c >> 3, ko = (c & 7) * 8;
    async_ld_b128(Qs + r * 72 + ko, Qb + (size_t)r * DMODEL + ko);
  }
  if (t < 128) { mi[t] = -3.0e38f; li[t] = 0.0f; }

  v8f oacc[2][2] = {};

  for (int s0 = 0; s0 < S; s0 += 64) {
    __syncthreads();
    const f16* Kb = Kv + ((size_t)(b * S + s0)) * DMODEL + h * DHEAD;
    const f16* Vb = Vv + ((size_t)(b * S + s0)) * DMODEL + h * DHEAD;
    for (int c = t; c < 64 * 8; c += 256) {
      int r = c >> 3, ko = (c & 7) * 8;
      async_ld_b128(Ks + r * 72 + ko, Kb + (size_t)r * DMODEL + ko);  // K: DMA to LDS
      v8h vv = *(const v8h*)(Vb + (size_t)r * DMODEL + ko);           // V: transpose via VGPRs
#pragma unroll
      for (int i = 0; i < 8; i++) Vt[(ko + i) * 72 + r] = vv[i];
    }
    wait_async_0();  // covers Q preload on first iteration too
    __syncthreads();

    // scores chunk = Q @ K^T  (128x64)
    v8f sacc[2][2] = {};
#pragma unroll
    for (int kb = 0; kb < 64; kb += 32) {
      v16h af[2], bfv[2];
#pragma unroll
      for (int tm = 0; tm < 2; tm++) af[tm] = ld_afrag(Qs, 72, wm * 32 + tm * 16 + ln, hi, kb);
#pragma unroll
      for (int tn = 0; tn < 2; tn++) bfv[tn] = ld_bfrag(Ks, 72, wn * 32 + tn * 16 + ln, hi, kb);
#pragma unroll
      for (int tm = 0; tm < 2; tm++)
#pragma unroll
        for (int tn = 0; tn < 2; tn++)
          sacc[tm][tn] = wmma_f16(af[tm], bfv[tn], sacc[tm][tn]);
    }
#pragma unroll
    for (int tm = 0; tm < 2; tm++)
#pragma unroll
      for (int tn = 0; tn < 2; tn++) {
        int n = wn * 32 + tn * 16 + ln;
#pragma unroll
        for (int r = 0; r < 8; r++) {
          int m = wm * 32 + tm * 16 + hi * 8 + r;
          Sc[m * 66 + n] = (sacc[tm][tn][r] * taub + delb) * scale;
        }
      }
    __syncthreads();

    // online softmax (one thread per query row)
    if (t < 128) {
      float mx = mi[t];
      for (int j = 0; j < 64; j++) { float v = Sc[t * 66 + j]; mx = v > mx ? v : mx; }
      float c = expf(mi[t] - mx);
      float s = 0.0f;
      for (int j = 0; j < 64; j++) {
        float p = expf(Sc[t * 66 + j] - mx);
        Ps[t * 72 + j] = (f16)p;
        s += p;
      }
      li[t] = li[t] * c + s;
      mi[t] = mx;
      cr[t] = c;
    }
    __syncthreads();

    // rescale running O, then O += P @ V
#pragma unroll
    for (int tm = 0; tm < 2; tm++)
#pragma unroll
      for (int tn = 0; tn < 2; tn++)
#pragma unroll
        for (int r = 0; r < 8; r++) {
          int m = wm * 32 + tm * 16 + hi * 8 + r;
          oacc[tm][tn][r] *= cr[m];
        }
#pragma unroll
    for (int kb = 0; kb < 64; kb += 32) {
      v16h af[2], bfv[2];
#pragma unroll
      for (int tm = 0; tm < 2; tm++) af[tm] = ld_afrag(Ps, 72, wm * 32 + tm * 16 + ln, hi, kb);
#pragma unroll
      for (int tn = 0; tn < 2; tn++) bfv[tn] = ld_bfrag(Vt, 72, wn * 32 + tn * 16 + ln, hi, kb);
#pragma unroll
      for (int tm = 0; tm < 2; tm++)
#pragma unroll
        for (int tn = 0; tn < 2; tn++)
          oacc[tm][tn] = wmma_f16(af[tm], bfv[tn], oacc[tm][tn]);
    }
  }
  __syncthreads();

  f16* Ob = O + ((size_t)b * PNUM) * DMODEL + h * DHEAD;
#pragma unroll
  for (int tm = 0; tm < 2; tm++)
#pragma unroll
    for (int tn = 0; tn < 2; tn++) {
      int n = wn * 32 + tn * 16 + ln;
#pragma unroll
      for (int r = 0; r < 8; r++) {
        int m = wm * 32 + tm * 16 + hi * 8 + r;
        Ob[(size_t)m * DMODEL + n] = (f16)(oacc[tm][tn][r] / li[m]);
      }
    }
}

// ---------------------------------------------------------------- glue kernels
__global__ __launch_bounds__(256) void norm_kernel(
    const float* __restrict__ x, float* __restrict__ xv,
    float* __restrict__ meanb, float* __restrict__ stdb) {
  __shared__ float red[256];
  int bc = blockIdx.x, b = bc >> 3, c = bc & 7;
  int t = threadIdx.x;
  float s = 0.0f;
  for (int i = t; i < SEQLEN; i += 256) s += x[((size_t)b * SEQLEN + i) * CIN + c];
  red[t] = s; __syncthreads();
  for (int o = 128; o > 0; o >>= 1) { if (t < o) red[t] += red[t + o]; __syncthreads(); }
  float mean = red[0] / (float)SEQLEN;
  __syncthreads();
  float v = 0.0f;
  for (int i = t; i < SEQLEN; i += 256) {
    float d = x[((size_t)b * SEQLEN + i) * CIN + c] - mean;
    v += d * d;
  }
  red[t] = v; __syncthreads();
  for (int o = 128; o > 0; o >>= 1) { if (t < o) red[t] += red[t + o]; __syncthreads(); }
  float stdv = sqrtf(red[0] / (float)SEQLEN + EPSV);
  if (t == 0) { meanb[bc] = mean; stdb[bc] = stdv; }
  float rs = 1.0f / stdv;
  for (int i = t; i < SEQLEN; i += 256)
    xv[(size_t)bc * SEQLEN + i] = (x[((size_t)b * SEQLEN + i) * CIN + c] - mean) * rs;
}

__global__ __launch_bounds__(256) void embed_kernel(
    const float* __restrict__ xv, const float* __restrict__ Wp,
    const float* __restrict__ raw, float* __restrict__ out32, f16* __restrict__ out16,
    int nvar, int cstart) {
  __shared__ float patch[PLEN];
  int bidx = blockIdx.x;
  int p = bidx % PNUM, v = (bidx / PNUM) % nvar, b = bidx / (PNUM * nvar);
  int t = threadIdx.x;
  if (t < PLEN) patch[t] = xv[((size_t)b * CIN + cstart + v) * SEQLEN + p * PLEN + t];
  __syncthreads();
  float sig = 1.0f / (1.0f + expf(-raw[p]));
  size_t row = ((size_t)b * nvar + v) * PNUM + p;
  for (int d = t; d < DMODEL; d += 256) {
    float s = 0.0f;
#pragma unroll
    for (int l = 0; l < PLEN; l++) s += Wp[d * PLEN + l] * patch[l];
    int j = d >> 1;
    float div = expf(-(float)(2 * j) * (9.210340371976184f / (float)DMODEL));
    float ang = (float)p * div;
    float pe = (d & 1) ? cosf(ang) : sinf(ang);
    float val = (s + pe) * sig;
    if (out32) out32[row * DMODEL + d] = val;
    out16[row * DMODEL + d] = (f16)val;
  }
}

__global__ __launch_bounds__(256) void patch_stats_kernel(
    const float* __restrict__ en, f16* __restrict__ mu16, f16* __restrict__ sd16) {
  int b = blockIdx.x, t = threadIdx.x;
  for (int d = t; d < DMODEL; d += 256) {
    float s = 0.0f;
    for (int p = 0; p < PNUM; p++) s += en[((size_t)b * PNUM + p) * DMODEL + d];
    float m = s / (float)PNUM;
    float v = 0.0f;
    for (int p = 0; p < PNUM; p++) {
      float x = en[((size_t)b * PNUM + p) * DMODEL + d] - m;
      v += x * x;
    }
    float sd = sqrtf(v / (float)(PNUM - 1));  // ddof=1
    mu16[(size_t)b * DMODEL + d] = (f16)m;
    sd16[(size_t)b * DMODEL + d] = (f16)sd;
  }
}

__global__ __launch_bounds__(256) void vecdot_kernel(
    const f16* __restrict__ Hm, const float* __restrict__ w2,
    const float* __restrict__ b2, float* __restrict__ out, int doExp) {
  __shared__ float red[256];
  int b = blockIdx.x, t = threadIdx.x;
  float s = 0.0f;
  for (int d = t; d < DMODEL; d += 256) s += (float)Hm[(size_t)b * DMODEL + d] * w2[d];
  red[t] = s; __syncthreads();
  for (int o = 128; o > 0; o >>= 1) { if (t < o) red[t] += red[t + o]; __syncthreads(); }
  if (t == 0) {
    float r = red[0] + b2[0];
    out[b] = doExp ? expf(r) : r;
  }
}

__global__ __launch_bounds__(256) void ln_kernel(
    float* __restrict__ x, f16* __restrict__ x16,
    const float* __restrict__ g, const float* __restrict__ b) {
  __shared__ float red[256];
  size_t row = blockIdx.x;
  int t = threadIdx.x;
  float v0 = x[row * DMODEL + t];
  float v1 = x[row * DMODEL + t + 256];
  float v2 = x[row * DMODEL + t + 512];
  red[t] = v0 + v1 + v2; __syncthreads();
  for (int o = 128; o > 0; o >>= 1) { if (t < o) red[t] += red[t + o]; __syncthreads(); }
  float mean = red[0] * (1.0f / 768.0f);
  __syncthreads();
  float d0 = v0 - mean, d1 = v1 - mean, d2 = v2 - mean;
  red[t] = d0 * d0 + d1 * d1 + d2 * d2; __syncthreads();
  for (int o = 128; o > 0; o >>= 1) { if (t < o) red[t] += red[t + o]; __syncthreads(); }
  float rstd = rsqrtf(red[0] * (1.0f / 768.0f) + EPSV);
  float o0 = d0 * rstd * g[t] + b[t];
  float o1 = d1 * rstd * g[t + 256] + b[t + 256];
  float o2 = d2 * rstd * g[t + 512] + b[t + 512];
  x[row * DMODEL + t] = o0; x[row * DMODEL + t + 256] = o1; x[row * DMODEL + t + 512] = o2;
  x16[row * DMODEL + t] = (f16)o0;
  x16[row * DMODEL + t + 256] = (f16)o1;
  x16[row * DMODEL + t + 512] = (f16)o2;
}

__global__ __launch_bounds__(256) void decomp_kernel(
    const float* __restrict__ xin, float* __restrict__ xo, f16* __restrict__ xo16) {
  int bp = blockIdx.x;
  int b = bp >> 7, p = bp & 127;
  int pm = p > 0 ? p - 1 : 0, pp = p < PNUM - 1 ? p + 1 : PNUM - 1;
  for (int d = threadIdx.x; d < DMODEL; d += 256) {
    float a = xin[((size_t)b * PNUM + pm) * DMODEL + d];
    float c = xin[((size_t)b * PNUM + p) * DMODEL + d];
    float e = xin[((size_t)b * PNUM + pp) * DMODEL + d];
    float v = c - (a + c + e) * (1.0f / 3.0f);
    xo[((size_t)b * PNUM + p) * DMODEL + d] = v;
    xo16[((size_t)b * PNUM + p) * DMODEL + d] = (f16)v;
  }
}

__global__ __launch_bounds__(256) void head_kernel(
    const float* __restrict__ x, const float* __restrict__ hw,
    const float* __restrict__ hb, const float* __restrict__ meanb,
    const float* __restrict__ stdb, float* __restrict__ out) {
  __shared__ float red[256];
  int tt = blockIdx.x, b = blockIdx.y, t = threadIdx.x;
  float s = 0.0f;
  for (int idx = t; idx < DMODEL * PNUM; idx += 256) {
    int d = idx >> 7, p = idx & 127;  // flat index = d*P + p
    s += x[((size_t)b * PNUM + p) * DMODEL + d] * hw[(size_t)tt * (DMODEL * PNUM) + idx];
  }
  red[t] = s; __syncthreads();
  for (int o = 128; o > 0; o >>= 1) { if (t < o) red[t] += red[t + o]; __syncthreads(); }
  if (t == 0) {
    float r = red[0] + hb[tt];
    out[(size_t)b * PRED + tt] = r * stdb[b * CIN + (CIN - 1)] + meanb[b * CIN + (CIN - 1)];
  }
}

__global__ __launch_bounds__(256) void f32tof16_kernel(
    const float* __restrict__ s, f16* __restrict__ d, int n) {
  for (int i = blockIdx.x * 256 + threadIdx.x; i < n; i += gridDim.x * 256)
    d[i] = (f16)s[i];
}

// ---------------------------------------------------------------- host side
struct LayerP {
  const float *sa_wq, *sa_bq, *sa_wk, *sa_bk, *sa_wv, *sa_bv, *sa_wo, *sa_bo;
  const float *ca_wq, *ca_bq, *ca_wk, *ca_bk, *ca_wv, *ca_bv, *ca_wo, *ca_bo;
  const float *wc1, *bc1, *wc2, *bc2, *g1, *b1, *g2, *b2;
};
struct ParamsP {
  const float *x_enc, *en_w, *en_raw, *ex_w, *ex_raw;
  const float *tau_w1, *tau_b1, *tau_w2, *tau_b2;
  const float *del_w1, *del_b1, *del_w2, *del_b2;
  const float *gf, *bf, *head_w, *head_b;
  LayerP L[NLAYER];
};

static void fill_insertion(void* const* d, ParamsP& P) {
  int i = 0;
  P.x_enc = (const float*)d[i++];
  P.en_w = (const float*)d[i++]; P.en_raw = (const float*)d[i++];
  P.ex_w = (const float*)d[i++]; P.ex_raw = (const float*)d[i++];
  P.tau_w1 = (const float*)d[i++]; P.tau_b1 = (const float*)d[i++];
  P.tau_w2 = (const float*)d[i++]; P.tau_b2 = (const float*)d[i++];
  P.del_w1 = (const float*)d[i++]; P.del_b1 = (const float*)d[i++];
  P.del_w2 = (const float*)d[i++]; P.del_b2 = (const float*)d[i++];
  for (int l = 0; l < NLAYER; l++) {
    LayerP& L = P.L[l];
    L.sa_wq = (const float*)d[i++]; L.sa_bq = (const float*)d[i++];
    L.sa_wk = (const float*)d[i++]; L.sa_bk = (const float*)d[i++];
    L.sa_wv = (const float*)d[i++]; L.sa_bv = (const float*)d[i++];
    L.sa_wo = (const float*)d[i++]; L.sa_bo = (const float*)d[i++];
    L.ca_wq = (const float*)d[i++]; L.ca_bq = (const float*)d[i++];
    L.ca_wk = (const float*)d[i++]; L.ca_bk = (const float*)d[i++];
    L.ca_wv = (const float*)d[i++]; L.ca_bv = (const float*)d[i++];
    L.ca_wo = (const float*)d[i++]; L.ca_bo = (const float*)d[i++];
    L.wc1 = (const float*)d[i++]; L.bc1 = (const float*)d[i++];
    L.wc2 = (const float*)d[i++]; L.bc2 = (const float*)d[i++];
    L.g1 = (const float*)d[i++]; L.b1 = (const float*)d[i++];
    L.g2 = (const float*)d[i++]; L.b2 = (const float*)d[i++];
  }
  P.gf = (const float*)d[i++]; P.bf = (const float*)d[i++];
  P.head_w = (const float*)d[i++]; P.head_b = (const float*)d[i++];
}

static void fill_sorted(void* const* d, ParamsP& P, int base, const float* xenc) {
  // jax tree-flatten order (dict keys sorted alphabetically)
  P.x_enc = xenc;
  int i = base;
  P.bf = (const float*)d[i++];
  P.del_b1 = (const float*)d[i++]; P.del_b2 = (const float*)d[i++];
  P.del_w1 = (const float*)d[i++]; P.del_w2 = (const float*)d[i++];
  P.en_raw = (const float*)d[i++]; P.en_w = (const float*)d[i++];
  P.ex_raw = (const float*)d[i++]; P.ex_w = (const float*)d[i++];
  P.gf = (const float*)d[i++];
  P.head_b = (const float*)d[i++]; P.head_w = (const float*)d[i++];
  for (int l = 0; l < NLAYER; l++) {
    LayerP& L = P.L[l];
    L.b1 = (const float*)d[i++]; L.b2 = (const float*)d[i++];
    L.bc1 = (const float*)d[i++]; L.bc2 = (const float*)d[i++];
    L.ca_bk = (const float*)d[i++]; L.ca_bo = (const float*)d[i++];
    L.ca_bq = (const float*)d[i++]; L.ca_bv = (const float*)d[i++];
    L.ca_wk = (const float*)d[i++]; L.ca_wo = (const float*)d[i++];
    L.ca_wq = (const float*)d[i++]; L.ca_wv = (const float*)d[i++];
    L.g1 = (const float*)d[i++]; L.g2 = (const float*)d[i++];
    L.sa_bk = (const float*)d[i++]; L.sa_bo = (const float*)d[i++];
    L.sa_bq = (const float*)d[i++]; L.sa_bv = (const float*)d[i++];
    L.sa_wk = (const float*)d[i++]; L.sa_wo = (const float*)d[i++];
    L.sa_wq = (const float*)d[i++]; L.sa_wv = (const float*)d[i++];
    L.wc1 = (const float*)d[i++]; L.wc2 = (const float*)d[i++];
  }
  P.tau_b1 = (const float*)d[i++]; P.tau_b2 = (const float*)d[i++];
  P.tau_w1 = (const float*)d[i++]; P.tau_w2 = (const float*)d[i++];
}

extern "C" void kernel_launch(void* const* d_in, const int* in_sizes, int n_in,
                              void* d_out, int out_size, void* d_ws, size_t ws_size,
                              hipStream_t stream) {
  if (n_in < 113) return;
  ParamsP P;
  if (in_sizes[0] == BATCH * SEQLEN * CIN) {
    if (in_sizes[1] == DMODEL * PLEN) fill_insertion(d_in, P);       // insertion order
    else fill_sorted(d_in, P, 1, (const float*)d_in[0]);             // x_enc first, sorted params
  } else {
    fill_sorted(d_in, P, 0, (const float*)d_in[n_in - 1]);           // sorted, x_enc last
  }

  // ---- workspace carve
  size_t off = 0;
  auto carve = [&](size_t bytes) -> char* {
    char* p = (char*)d_ws + off;
    off = (off + bytes + 255) & ~(size_t)255;
    return p;
  };
  const size_t TOK = (size_t)BATCH * PNUM;           // 8192
  const size_t XTOK = (size_t)BATCH * NEXO * PNUM;   // 57344
  float* xv = (float*)carve((size_t)BATCH * CIN * SEQLEN * 4);
  float* meanb = (float*)carve(BATCH * CIN * 4);
  float* stdb = (float*)carve(BATCH * CIN * 4);
  float* cur32 = (float*)carve(TOK * DMODEL * 4);
  float* alt32 = (float*)carve(TOK * DMODEL * 4);
  f16* cur16 = (f16*)carve(TOK * DMODEL * 2);
  f16* alt16 = (f16*)carve(TOK * DMODEL * 2);
  f16* ex16 = (f16*)carve(XTOK * DMODEL * 2);
  f16* q16 = (f16*)carve(TOK * DMODEL * 2);
  f16* k16s = (f16*)carve(TOK * DMODEL * 2);
  f16* v16s = (f16*)carve(TOK * DMODEL * 2);
  f16* o16 = (f16*)carve(TOK * DMODEL * 2);
  f16* kx16 = (f16*)carve(XTOK * DMODEL * 2);
  f16* vx16 = (f16*)carve(XTOK * DMODEL * 2);
  f16* h16 = (f16*)carve(TOK * DFFN * 2);
  f16* mu16 = (f16*)carve(BATCH * DMODEL * 2);
  f16* sd16 = (f16*)carve(BATCH * DMODEL * 2);
  f16* th16 = (f16*)carve(BATCH * DMODEL * 2);
  f16* dh16 = (f16*)carve(BATCH * DMODEL * 2);
  float* taub = (float*)carve(BATCH * 4);
  float* deltab = (float*)carve(BATCH * 4);
  f16* tw1h = (f16*)carve((size_t)DMODEL * DMODEL * 2);
  f16* dw1h = (f16*)carve((size_t)DMODEL * DMODEL * 2);
  f16* wh[NLAYER][10];
  for (int l = 0; l < NLAYER; l++) {
    for (int k = 0; k < 8; k++) wh[l][k] = (f16*)carve((size_t)DMODEL * DMODEL * 2);
    wh[l][8] = (f16*)carve((size_t)DFFN * DMODEL * 2);
    wh[l][9] = (f16*)carve((size_t)DMODEL * DFFN * 2);
  }
  if (off > ws_size) return;  // not enough scratch; bail without launching

  auto conv = [&](const float* s, f16* d, int n) {
    f32tof16_kernel<<<1024, 256, 0, stream>>>(s, d, n);
  };
  auto gemm = [&](const f16* A, const f16* W, const float* bias, const float* resid,
                  float* o32, f16* o16p, int M, int N, int K, int flags) {
    dim3 g((N + 63) / 64, (M + 127) / 128);
    if ((M & 127) == 0 && (N & 63) == 0)
      gemm_wmma_kernel<false><<<g, 256, 0, stream>>>(A, W, bias, resid, o32, o16p, M, N, K, flags);
    else
      gemm_wmma_kernel<true><<<g, 256, 0, stream>>>(A, W, bias, resid, o32, o16p, M, N, K, flags);
  };

  // ---- pre-convert all GEMM weights to f16 (recomputed every call: deterministic)
  conv(P.tau_w1, tw1h, DMODEL * DMODEL);
  conv(P.del_w1, dw1h, DMODEL * DMODEL);
  for (int l = 0; l < NLAYER; l++) {
    const LayerP& L = P.L[l];
    conv(L.sa_wq, wh[l][0], DMODEL * DMODEL);
    conv(L.sa_wk, wh[l][1], DMODEL * DMODEL);
    conv(L.sa_wv, wh[l][2], DMODEL * DMODEL);
    conv(L.sa_wo, wh[l][3], DMODEL * DMODEL);
    conv(L.ca_wq, wh[l][4], DMODEL * DMODEL);
    conv(L.ca_wk, wh[l][5], DMODEL * DMODEL);
    conv(L.ca_wv, wh[l][6], DMODEL * DMODEL);
    conv(L.ca_wo, wh[l][7], DMODEL * DMODEL);
    conv(L.wc1, wh[l][8], DFFN * DMODEL);
    conv(L.wc2, wh[l][9], DMODEL * DFFN);
  }

  // ---- instance norm, embeddings
  norm_kernel<<<BATCH * CIN, 256, 0, stream>>>(P.x_enc, xv, meanb, stdb);
  embed_kernel<<<BATCH * PNUM, 256, 0, stream>>>(xv, P.en_w, P.en_raw, cur32, cur16, 1, CIN - 1);
  embed_kernel<<<BATCH * NEXO * PNUM, 256, 0, stream>>>(xv, P.ex_w, P.ex_raw, nullptr, ex16, NEXO, 0);

  // ---- tau / delta learners
  patch_stats_kernel<<<BATCH, 256, 0, stream>>>(cur32, mu16, sd16);
  gemm(sd16, tw1h, P.tau_b1, nullptr, nullptr, th16, BATCH, DMODEL, DMODEL,
       GF_BIAS | GF_RELU | GF_O16);
  gemm(mu16, dw1h, P.del_b1, nullptr, nullptr, dh16, BATCH, DMODEL, DMODEL,
       GF_BIAS | GF_RELU | GF_O16);
  vecdot_kernel<<<BATCH, 256, 0, stream>>>(th16, P.tau_w2, P.tau_b2, taub, 1);
  vecdot_kernel<<<BATCH, 256, 0, stream>>>(dh16, P.del_w2, P.del_b2, deltab, 0);

  // ---- encoder layers
  for (int l = 0; l < NLAYER; l++) {
    const LayerP& L = P.L[l];
    // self-attention (tau/delta scaled)
    gemm(cur16, wh[l][0], L.sa_bq, nullptr, nullptr, q16, TOK, DMODEL, DMODEL, GF_BIAS | GF_O16);
    gemm(cur16, wh[l][1], L.sa_bk, nullptr, nullptr, k16s, TOK, DMODEL, DMODEL, GF_BIAS | GF_O16);
    gemm(cur16, wh[l][2], L.sa_bv, nullptr, nullptr, v16s, TOK, DMODEL, DMODEL, GF_BIAS | GF_O16);
    attn_wmma_kernel<<<dim3(NHEAD, BATCH), 256, ATTN_SMEM, stream>>>(
        q16, k16s, v16s, o16, taub, deltab, PNUM);
    gemm(o16, wh[l][3], L.sa_bo, cur32, alt32, alt16, TOK, DMODEL, DMODEL,
         GF_BIAS | GF_RES | GF_O32 | GF_O16);
    { float* tf = cur32; cur32 = alt32; alt32 = tf; f16* th = cur16; cur16 = alt16; alt16 = th; }
    // series decomp (trend accumulator is dead in the reference -> skipped)
    decomp_kernel<<<BATCH * PNUM, 256, 0, stream>>>(cur32, alt32, alt16);
    { float* tf = cur32; cur32 = alt32; alt32 = tf; f16* th = cur16; cur16 = alt16; alt16 = th; }
    // cross-attention with exogenous tokens
    gemm(cur16, wh[l][4], L.ca_bq, nullptr, nullptr, q16, TOK, DMODEL, DMODEL, GF_BIAS | GF_O16);
    gemm(ex16, wh[l][5], L.ca_bk, nullptr, nullptr, kx16, XTOK, DMODEL, DMODEL, GF_BIAS | GF_O16);
    gemm(ex16, wh[l][6], L.ca_bv, nullptr, nullptr, vx16, XTOK, DMODEL, DMODEL, GF_BIAS | GF_O16);
    attn_wmma_kernel<<<dim3(NHEAD, BATCH), 256, ATTN_SMEM, stream>>>(
        q16, kx16, vx16, o16, nullptr, nullptr, NEXO * PNUM);
    gemm(o16, wh[l][7], L.ca_bo, cur32, alt32, alt16, TOK, DMODEL, DMODEL,
         GF_BIAS | GF_RES | GF_O32 | GF_O16);
    { float* tf = cur32; cur32 = alt32; alt32 = tf; f16* th = cur16; cur16 = alt16; alt16 = th; }
    // norm1 + FFN + residual
    ln_kernel<<<(int)TOK, 256, 0, stream>>>(cur32, cur16, L.g1, L.b1);
    gemm(cur16, wh[l][8], L.bc1, nullptr, nullptr, h16, TOK, DFFN, DMODEL,
         GF_BIAS | GF_GELU | GF_O16);
    gemm(h16, wh[l][9], L.bc2, cur32, alt32, alt16, TOK, DMODEL, DFFN,
         GF_BIAS | GF_RES | GF_O32 | GF_O16);
    { float* tf = cur32; cur32 = alt32; alt32 = tf; f16* th = cur16; cur16 = alt16; alt16 = th; }
    // decomp + norm2
    decomp_kernel<<<BATCH * PNUM, 256, 0, stream>>>(cur32, alt32, alt16);
    { float* tf = cur32; cur32 = alt32; alt32 = tf; f16* th = cur16; cur16 = alt16; alt16 = th; }
    ln_kernel<<<(int)TOK, 256, 0, stream>>>(cur32, cur16, L.g2, L.b2);
  }

  // ---- final LN + head + de-normalization
  ln_kernel<<<(int)TOK, 256, 0, stream>>>(cur32, cur16, P.gf, P.bf);
  head_kernel<<<dim3(PRED, BATCH), 256, 0, stream>>>(cur32, P.head_w, P.head_b, meanb, stdb,
                                                     (float*)d_out);
}
